// MultiHeadAttention_90683939488307
// MI455X (gfx1250) — matrix operand order repaired and checked by reading, not compiled, not run on hardware
//
#include <hip/hip_runtime.h>
#include <hip/hip_bf16.h>

// ---------- types ----------
typedef __bf16 bf16;
typedef __attribute__((ext_vector_type(16))) __bf16 v16bf;
typedef __attribute__((ext_vector_type(8)))  __bf16 v8bf;
typedef __attribute__((ext_vector_type(2)))  __bf16 v2bf;
typedef __attribute__((ext_vector_type(8)))  float  v8f;
typedef __attribute__((ext_vector_type(4)))  int    v4i;

#define AS1 __attribute__((address_space(1)))
#define AS3 __attribute__((address_space(3)))

// Async global->LDS copies (CDNA5, ASYNCcnt-tracked). Guarded so the kernel
// still compiles if this toolchain lacks the builtins.
#if defined(__gfx1250__) &&                                                   \
    __has_builtin(__builtin_amdgcn_global_load_async_to_lds_b128) &&          \
    __has_builtin(__builtin_amdgcn_s_wait_asynccnt)
#define HAVE_ASYNC_LDS 1
#endif

static __device__ __forceinline__ void async_copy_b128(const bf16* g, bf16* l) {
#ifdef HAVE_ASYNC_LDS
  __builtin_amdgcn_global_load_async_to_lds_b128((AS1 v4i*)(void*)g,
                                                 (AS3 v4i*)(void*)l, 0, 0);
#else
  *(v8bf*)l = *(const v8bf*)g;
#endif
}
static __device__ __forceinline__ void async_wait_all() {
#ifdef HAVE_ASYNC_LDS
  __builtin_amdgcn_s_wait_asynccnt(0);
#endif
}

static __device__ __forceinline__ v8f wmma_bf16(v16bf a, v16bf b, v8f c) {
  return __builtin_amdgcn_wmma_f32_16x16x32_bf16(false, a, false, b, (short)0, c,
                                                 false, false);
}

// A-fragment (16x32): lane m = lane%16, half = lane/16; elem j -> K = j + 16*half
// (two contiguous 16B runs at K = 16*half and K = 16*half + 8).
static __device__ __forceinline__ v16bf load_a_frag(const bf16* rowptr, int half) {
  v8bf lo = *(const v8bf*)(rowptr + 16 * half);
  v8bf hi = *(const v8bf*)(rowptr + 16 * half + 8);
  v16bf r;
#pragma unroll
  for (int i = 0; i < 8; ++i) { r[i] = lo[i]; r[i + 8] = hi[i]; }
  return r;
}
// B-fragment: lane n = lane%16, half = lane/16; elem j -> K = j + 16*half
// => 16 contiguous bf16 (caller passes pointer at K = 16*half).
static __device__ __forceinline__ v16bf load_b_frag(const bf16* p) {
  v8bf lo = *(const v8bf*)(p);
  v8bf hi = *(const v8bf*)(p + 8);
  v16bf r;
#pragma unroll
  for (int i = 0; i < 8; ++i) { r[i] = lo[i]; r[i + 8] = hi[i]; }
  return r;
}

#define N_HEAD 16
#define SEQ    2048
#define DMODEL 1024
#define HDIM   64

// =====================================================================
// Kernel 1: QKV projection. X[M,1024] f32 @ Wqkv[1024,3072] f32 + b.
// Block tile 128x128, 8 waves (4x2), wave tile 32x64 => 8 WMMA / K-step.
// Output scattered as bf16 into Q/K/V workspaces laid out [B,H,S,hd].
// =====================================================================
__global__ __launch_bounds__(256) void qkv_gemm_kernel(
    const float* __restrict__ X, const float* __restrict__ W,
    const float* __restrict__ bias,
    bf16* __restrict__ q_ws, bf16* __restrict__ k_ws, bf16* __restrict__ v_ws) {
  const int K = DMODEL, N = 3 * DMODEL;
  __shared__ bf16 As[128 * 32];   // [m][k]
  __shared__ bf16 Bs[128 * 32];   // [n][k]

  const int bn = blockIdx.x;      // 24
  const int bm = blockIdx.y;      // 64
  const int tid = threadIdx.x;
  const int wave = tid >> 5, lane = tid & 31;
  const int wm = wave >> 1, wn = wave & 1;        // 4 x 2 wave grid
  const int lhalf = lane >> 4, lmod = lane & 15;

  v8f acc[2][4] = {};
  for (int k0 = 0; k0 < K; k0 += 32) {
    // X tile 128x32: packed bf16x2 stores (2048 pairs / 256 thr = 8 each)
#pragma unroll
    for (int i = 0; i < 8; ++i) {
      int idx = tid + i * 256;
      int r = idx >> 4, t = idx & 15;             // 16 k-pairs per row
      float2 xv = *(const float2*)&X[(size_t)(bm * 128 + r) * K + k0 + 2 * t];
      v2bf pk = {(bf16)xv.x, (bf16)xv.y};
      *(v2bf*)&As[r * 32 + 2 * t] = pk;
    }
    // W tile 32x128 -> Bs[n][k] transposed, packed pairs along k
#pragma unroll
    for (int i = 0; i < 8; ++i) {
      int idx = tid + i * 256;
      int t = idx >> 7, n = idx & 127;            // consecutive threads -> consecutive n
      const float* wp = &W[(size_t)(k0 + 2 * t) * N + bn * 128 + n];
      v2bf pk = {(bf16)wp[0], (bf16)wp[N]};
      *(v2bf*)&Bs[n * 32 + 2 * t] = pk;
    }
    __syncthreads();

    v16bf afr[2], bfr[4];
#pragma unroll
    for (int mt = 0; mt < 2; ++mt)
      afr[mt] = load_a_frag(&As[(wm * 32 + mt * 16 + lmod) * 32], lhalf);
#pragma unroll
    for (int nt = 0; nt < 4; ++nt)
      bfr[nt] = load_b_frag(&Bs[(wn * 64 + nt * 16 + lmod) * 32 + 16 * lhalf]);
#pragma unroll
    for (int mt = 0; mt < 2; ++mt)
#pragma unroll
      for (int nt = 0; nt < 4; ++nt)
        acc[mt][nt] = wmma_bf16(afr[mt], bfr[nt], acc[mt][nt]);
    __syncthreads();
  }

  // Scatter-store. C layout: lane n = lmod; vgpr v -> m = v + 8*lhalf.
#pragma unroll
  for (int mt = 0; mt < 2; ++mt) {
#pragma unroll
    for (int nt = 0; nt < 4; ++nt) {
      int row0 = bm * 128 + wm * 32 + mt * 16 + 8 * lhalf;   // aligned to 8
      int col = bn * 128 + wn * 64 + nt * 16 + lmod;
      float bia = bias[col];
      int t = col >> 10, ch = col & 1023;
      int h = ch >> 6, d = ch & 63;
      int b = row0 >> 11, s0 = row0 & (SEQ - 1);             // same b for v=0..7
      bf16* dst = (t == 0) ? q_ws : (t == 1) ? k_ws : v_ws;
      size_t base = (((size_t)(b * N_HEAD + h)) * SEQ + s0) * HDIM + d;
#pragma unroll
      for (int v = 0; v < 8; ++v)
        dst[base + (size_t)v * HDIM] = (bf16)(acc[mt][nt][v] + bia);
    }
  }
}

// =====================================================================
// Kernel 2: causal flash attention. Block = (b,h, 128-query chunk), 8 waves,
// each wave owns a 16-row query tile. Key loop in steps of 64:
// 8 score WMMAs + 8 PV WMMAs per step. K tile via async global->LDS.
// =====================================================================
__global__ __launch_bounds__(256) void attn_kernel(
    const bf16* __restrict__ q_ws, const bf16* __restrict__ k_ws,
    const bf16* __restrict__ v_ws, bf16* __restrict__ a_ws) {
  __shared__ bf16 Ks[64 * 64];        // [key][d]
  __shared__ bf16 Vs[64 * 64];        // [d][key]   (transposed)
  __shared__ bf16 Ps[8][16 * 64];     // per-wave P buffer [m][key]

  const int qc = blockIdx.x & 15;     // S/128 = 16 chunks
  const int bh = blockIdx.x >> 4;     // b*H + h
  const int b = bh >> 4, h = bh & (N_HEAD - 1);
  const int tid = threadIdx.x, wave = tid >> 5, lane = tid & 31;
  const int lhalf = lane >> 4, lmod = lane & 15;
  const int q0 = qc * 128 + wave * 16;

  const bf16* Qbase = q_ws + ((size_t)bh * SEQ + q0) * HDIM;
  v16bf qf[2];
  qf[0] = load_a_frag(Qbase + lmod * HDIM, lhalf);
  qf[1] = load_a_frag(Qbase + lmod * HDIM + 32, lhalf);

  v8f acc[4] = {};
  float mrow[8], lrow[8];
#pragma unroll
  for (int v = 0; v < 8; ++v) { mrow[v] = -1e30f; lrow[v] = 0.0f; }

  const int nkb = (qc + 1) * 2;       // 64-key blocks (uniform per block)
  for (int jb = 0; jb < nkb; ++jb) {
    const int kbase = jb * 64;
    // K tile 64x64 bf16 = 512 x 16B chunks, async copy (2 per thread)
#pragma unroll
    for (int c = 0; c < 2; ++c) {
      int idx = tid + c * 256;
      int r = idx >> 3, off = (idx & 7) * 8;     // 8 chunks per 128B row
      async_copy_b128(&k_ws[((size_t)bh * SEQ + kbase + r) * HDIM + off],
                      &Ks[r * 64 + off]);
    }
    // V tile transposed into Vs[d][key], packed key-pairs (b32 stores)
#pragma unroll
    for (int i = 0; i < 8; ++i) {
      int idx = tid + i * 256;                   // 2048 pairs
      int d = idx >> 5, t = idx & 31;            // keys 2t, 2t+1
      const bf16* vp = &v_ws[((size_t)bh * SEQ + kbase + 2 * t) * HDIM + d];
      v2bf pk = {vp[0], vp[HDIM]};
      *(v2bf*)&Vs[d * 64 + 2 * t] = pk;
    }
    async_wait_all();
    __syncthreads();

    // Scores: four 16-key C tiles, hd=64 reduction (2 WMMAs each)
    v8f sc[4] = {};
#pragma unroll
    for (int ct = 0; ct < 4; ++ct) {
      v16bf kb0 = load_b_frag(&Ks[(ct * 16 + lmod) * 64 + 16 * lhalf]);
      v16bf kb1 = load_b_frag(&Ks[(ct * 16 + lmod) * 64 + 32 + 16 * lhalf]);
      sc[ct] = wmma_bf16(qf[0], kb0, sc[ct]);
      sc[ct] = wmma_bf16(qf[1], kb1, sc[ct]);
    }

    // Online softmax (row = v + 8*lhalf lives within one 16-lane half)
    float p[4][8];
#pragma unroll
    for (int v = 0; v < 8; ++v) {
      int qrow = q0 + v + 8 * lhalf;
      float sv[4];
      float m = -1e30f;
#pragma unroll
      for (int ct = 0; ct < 4; ++ct) {
        int key = kbase + ct * 16 + lmod;
        float s = (key <= qrow) ? sc[ct][v] * 0.125f : -1e9f;
        sv[ct] = s;
        m = fmaxf(m, s);
      }
      m = fmaxf(m, __shfl_xor(m, 1, 32));
      m = fmaxf(m, __shfl_xor(m, 2, 32));
      m = fmaxf(m, __shfl_xor(m, 4, 32));
      m = fmaxf(m, __shfl_xor(m, 8, 32));
      float mnew = fmaxf(mrow[v], m);
      float alpha = __expf(mrow[v] - mnew);
      float rs = 0.0f;
#pragma unroll
      for (int ct = 0; ct < 4; ++ct) {
        float e = (sv[ct] > -1e8f) ? __expf(sv[ct] - mnew) : 0.0f;
        p[ct][v] = e;
        rs += e;
      }
      rs += __shfl_xor(rs, 1, 32);
      rs += __shfl_xor(rs, 2, 32);
      rs += __shfl_xor(rs, 4, 32);
      rs += __shfl_xor(rs, 8, 32);
      lrow[v] = lrow[v] * alpha + rs;
      mrow[v] = mnew;
#pragma unroll
      for (int nt = 0; nt < 4; ++nt) acc[nt][v] *= alpha;
    }

    // Transpose P (16x64) into A-fragment layout via wave-private LDS
    bf16* Pw = &Ps[wave][0];
#pragma unroll
    for (int ct = 0; ct < 4; ++ct)
#pragma unroll
      for (int v = 0; v < 8; ++v)
        Pw[(v + 8 * lhalf) * 64 + ct * 16 + lmod] = (bf16)p[ct][v];
    v16bf pf0 = load_a_frag(&Pw[lmod * 64], lhalf);
    v16bf pf1 = load_a_frag(&Pw[lmod * 64 + 32], lhalf);

    // PV: A = P (16x64 as two K32 slices), B = V slices (32 keys x 16 d)
#pragma unroll
    for (int nt = 0; nt < 4; ++nt) {
      v16bf vb0 = load_b_frag(&Vs[(nt * 16 + lmod) * 64 + 16 * lhalf]);
      v16bf vb1 = load_b_frag(&Vs[(nt * 16 + lmod) * 64 + 32 + 16 * lhalf]);
      acc[nt] = wmma_bf16(pf0, vb0, acc[nt]);
      acc[nt] = wmma_bf16(pf1, vb1, acc[nt]);
    }
    __syncthreads();
  }

  // Finalize: divide by l, store merged-heads bf16 [B,S,D]
#pragma unroll
  for (int v = 0; v < 8; ++v) {
    float inv = 1.0f / lrow[v];
    int s = q0 + v + 8 * lhalf;
#pragma unroll
    for (int nt = 0; nt < 4; ++nt) {
      int d = nt * 16 + lmod;
      a_ws[(((size_t)(b * SEQ + s)) * N_HEAD + h) * HDIM + d] =
          (bf16)(acc[nt][v] * inv);
    }
  }
}

// =====================================================================
// Kernel 3: output projection. A[M,1024] bf16 @ Wo[1024,1024] f32 + bo -> f32.
// Same 128x128 tiling; A tile copied via async global->LDS (already bf16).
// =====================================================================
__global__ __launch_bounds__(256) void out_gemm_kernel(
    const bf16* __restrict__ A, const float* __restrict__ W,
    const float* __restrict__ bias, float* __restrict__ out) {
  const int K = DMODEL, N = DMODEL;
  __shared__ bf16 As[128 * 32];
  __shared__ bf16 Bs[128 * 32];

  const int bn = blockIdx.x;      // 8
  const int bm = blockIdx.y;      // 64
  const int tid = threadIdx.x;
  const int wave = tid >> 5, lane = tid & 31;
  const int wm = wave >> 1, wn = wave & 1;
  const int lhalf = lane >> 4, lmod = lane & 15;

  v8f acc[2][4] = {};
  for (int k0 = 0; k0 < K; k0 += 32) {
    // A tile 128x32 bf16 = 512 x 16B chunks, async copy (2 per thread)
#pragma unroll
    for (int c = 0; c < 2; ++c) {
      int idx = tid + c * 256;
      int r = idx >> 2, off = (idx & 3) * 8;     // 4 chunks per 64B row
      async_copy_b128(&A[(size_t)(bm * 128 + r) * K + k0 + off],
                      &As[r * 32 + off]);
    }
    // W tile 32x128 -> Bs[n][k], packed pairs along k
#pragma unroll
    for (int i = 0; i < 8; ++i) {
      int idx = tid + i * 256;
      int t = idx >> 7, n = idx & 127;
      const float* wp = &W[(size_t)(k0 + 2 * t) * N + bn * 128 + n];
      v2bf pk = {(bf16)wp[0], (bf16)wp[N]};
      *(v2bf*)&Bs[n * 32 + 2 * t] = pk;
    }
    async_wait_all();
    __syncthreads();

    v16bf afr[2], bfr[4];
#pragma unroll
    for (int mt = 0; mt < 2; ++mt)
      afr[mt] = load_a_frag(&As[(wm * 32 + mt * 16 + lmod) * 32], lhalf);
#pragma unroll
    for (int nt = 0; nt < 4; ++nt)
      bfr[nt] = load_b_frag(&Bs[(wn * 64 + nt * 16 + lmod) * 32 + 16 * lhalf]);
#pragma unroll
    for (int mt = 0; mt < 2; ++mt)
#pragma unroll
      for (int nt = 0; nt < 4; ++nt)
        acc[mt][nt] = wmma_bf16(afr[mt], bfr[nt], acc[mt][nt]);
    __syncthreads();
  }

#pragma unroll
  for (int mt = 0; mt < 2; ++mt)
#pragma unroll
    for (int nt = 0; nt < 4; ++nt) {
      int row0 = bm * 128 + wm * 32 + mt * 16 + 8 * lhalf;
      int col = bn * 128 + wn * 64 + nt * 16 + lmod;
      float bia = bias[col];
      size_t base = (size_t)row0 * N + col;
#pragma unroll
      for (int v = 0; v < 8; ++v)
        out[base + (size_t)v * N] = acc[mt][nt][v] + bia;
    }
}

// Pass-through of string_len into the tail of d_out (tuple output)
__global__ void copy_len_kernel(const int* __restrict__ slen,
                                int* __restrict__ dst, int n) {
  int i = threadIdx.x;
  if (i < n) dst[i] = slen[i];
}

extern "C" void kernel_launch(void* const* d_in, const int* in_sizes, int n_in,
                              void* d_out, int out_size, void* d_ws, size_t ws_size,
                              hipStream_t stream) {
  const float* X    = (const float*)d_in[0];   // [B,S,D] f32
  const int*   slen = (const int*)d_in[1];     // [B] i32
  const float* Wqkv = (const float*)d_in[2];   // [D,3D]
  const float* bqkv = (const float*)d_in[3];   // [3D]
  const float* Wo   = (const float*)d_in[4];   // [D,D]
  const float* bo   = (const float*)d_in[5];   // [D]
  float* out = (float*)d_out;

  const int B = 4, S = SEQ, D = DMODEL, H = N_HEAD, hd = HDIM;
  const int M = B * S;                         // 8192
  const size_t per = (size_t)B * H * S * hd;   // 8,388,608 elems

  bf16* q_ws = (bf16*)d_ws;
  bf16* k_ws = q_ws + per;
  bf16* v_ws = k_ws + per;
  bf16* a_ws = v_ws + per;

  dim3 g1(3 * D / 128, M / 128);
  qkv_gemm_kernel<<<g1, 256, 0, stream>>>(X, Wqkv, bqkv, q_ws, k_ws, v_ws);

  attn_kernel<<<B * H * (S / 128), 256, 0, stream>>>(q_ws, k_ws, v_ws, a_ws);

  dim3 g3(D / 128, M / 128);
  out_gemm_kernel<<<g3, 256, 0, stream>>>(a_ws, Wo, bo, out);

  copy_len_kernel<<<1, 32, 0, stream>>>(slen, (int*)d_out + (size_t)M * D,
                                        in_sizes[1]);
}
